// Attention_87875030876861
// MI455X (gfx1250) — matrix operand order repaired and checked by reading, not compiled, not run on hardware
//
#include <hip/hip_runtime.h>

// ---------------------------------------------------------------------------
// MHA forward for B=32, N=512, C=1024, H=16, HD=64 on gfx1250 (wave32, WMMA).
// bf16 tensor-core path with fp32 accumulation + async global->LDS staging.
// ---------------------------------------------------------------------------

#define B_ 32
#define N_ 512
#define C_ 1024
#define H_ 16
#define HD_ 64
#define M_ (B_ * N_)        // 16384 rows of x
#define C3_ (3 * C_)        // 3072
#define SCALE_ 0.125f       // 64^-0.5
#define LOG2E_ 1.4426950408889634f

typedef __bf16 bf16_t;
typedef __attribute__((ext_vector_type(16))) __bf16 v16bf;
typedef __attribute__((ext_vector_type(8)))  __bf16 v8bf;
typedef __attribute__((ext_vector_type(8)))  float  v8f;

union FragBF { v16bf v; v8bf h[2]; };

__device__ __forceinline__ bf16_t f2bf(float f) {
  unsigned u = __builtin_bit_cast(unsigned, f);
  unsigned r = u + 0x7FFFu + ((u >> 16) & 1u);   // round-to-nearest-even
  unsigned short hs = (unsigned short)(r >> 16);
  return __builtin_bit_cast(__bf16, hs);
}

#if __has_builtin(__builtin_amdgcn_exp2f)
#define EXP2F(x) __builtin_amdgcn_exp2f(x)
#else
#define EXP2F(x) exp2f(x)
#endif

__device__ __forceinline__ void wait_ds0() {
#if __has_builtin(__builtin_amdgcn_s_wait_dscnt)
  __builtin_amdgcn_s_wait_dscnt(0);
#else
  asm volatile("s_wait_dscnt 0" ::: "memory");
#endif
}

__device__ __forceinline__ void wait_async0() {
#if __has_builtin(__builtin_amdgcn_s_wait_asynccnt)
  __builtin_amdgcn_s_wait_asynccnt(0);
#else
  asm volatile("s_wait_asynccnt 0" ::: "memory");
#endif
}

// Async copy: 32 lanes x 16B = 512B per issue, global -> LDS (ASYNCcnt).
// lds_off = byte offset within the workgroup LDS allocation (flat-LDS rule:
// low 32 bits of a generic pointer to __shared__ are the LDS offset).
__device__ __forceinline__ void async_copy_b128(unsigned lds_off, const bf16_t* g) {
  asm volatile("global_load_async_to_lds_b128 %0, %1, off"
               :: "v"(lds_off), "v"((unsigned long long)(size_t)g)
               : "memory");
}

// D = A(16x32 bf16) x B(32x16 bf16) + C, fp32 accum
__device__ __forceinline__ v8f wmma_bf16(v16bf a, v16bf b, v8f c) {
  return __builtin_amdgcn_wmma_f32_16x16x32_bf16(
      false, a, false, b, (short)0, c, false, false);
}

// A-operand frag: 16x32 tile, rowptr = &A[row16][k0] (row-major, row=lane&15).
// lo lanes hold K {0..7,16..23}; hi lanes K {8..15,24..31}.
__device__ __forceinline__ v16bf load_a_frag(const bf16_t* rowptr, int half) {
  FragBF u;
  u.h[0] = *(const v8bf*)(rowptr + half * 8);
  u.h[1] = *(const v8bf*)(rowptr + 16 + half * 8);
  return u.v;
}

// B-operand frag (column-major B): colptr = &Bn[n16][k0] where Bn is N x K
// row-major. lo lanes hold K {0..7,16..23}; hi lanes K {8..15,24..31}.
__device__ __forceinline__ v16bf load_b_frag(const bf16_t* colptr, int half) {
  FragBF u;
  u.h[0] = *(const v8bf*)(colptr + half * 8);
  u.h[1] = *(const v8bf*)(colptr + 16 + half * 8);
  return u.v;
}

// ---------------------------------------------------------------------------
__global__ __launch_bounds__(256) void cvt_f32_bf16_kernel(
    const float* __restrict__ in, bf16_t* __restrict__ out, int n) {
  int i = blockIdx.x * blockDim.x + threadIdx.x;
  if (i < n) out[i] = f2bf(in[i]);
}

// ---------------------------------------------------------------------------
// QKV GEMM: [16384,1024] x W_qkv^T -> scatter into Qs/Ks ([B,H,N,HD]) and
// Vt ([B,H,HD,N]). One wave per 32x64 output tile.
__global__ __launch_bounds__(256) void qkv_gemm_kernel(
    const bf16_t* __restrict__ Xb, const bf16_t* __restrict__ Wq,
    bf16_t* __restrict__ Qs, bf16_t* __restrict__ Ks, bf16_t* __restrict__ Vt) {
  const int lane = threadIdx.x & 31, wave = threadIdx.x >> 5;
  const int half = lane >> 4, l16 = lane & 15;
  const int wg = blockIdx.x * 8 + wave;           // 0..24575
  const int mt = wg / (C3_ / 64), nt = wg % (C3_ / 64);
  const int m0 = mt * 32, n0 = nt * 64;

  v8f c[2][4];
#pragma unroll
  for (int i = 0; i < 2; ++i)
#pragma unroll
    for (int j = 0; j < 4; ++j) c[i][j] = v8f{};

  const bf16_t* arow[2];
  arow[0] = Xb + (size_t)(m0 + l16) * C_;
  arow[1] = Xb + (size_t)(m0 + 16 + l16) * C_;
  const bf16_t* brow[4];
#pragma unroll
  for (int j = 0; j < 4; ++j) brow[j] = Wq + (size_t)(n0 + j * 16 + l16) * C_;

  for (int k0 = 0; k0 < C_; k0 += 32) {
    v16bf a0 = load_a_frag(arow[0] + k0, half);
    v16bf a1 = load_a_frag(arow[1] + k0, half);
#pragma unroll
    for (int j = 0; j < 4; ++j) {
      v16bf bb = load_b_frag(brow[j] + k0, half);
      c[0][j] = wmma_bf16(a0, bb, c[0][j]);
      c[1][j] = wmma_bf16(a1, bb, c[1][j]);
    }
  }

  const int which = n0 >> 10;          // 0=Q 1=K 2=V (constant per tile)
  const int h = (n0 >> 6) & (H_ - 1);  // head (constant per 64-wide tile)
#pragma unroll
  for (int i = 0; i < 2; ++i)
#pragma unroll
    for (int j = 0; j < 4; ++j)
#pragma unroll
      for (int r = 0; r < 8; ++r) {
        int m = m0 + i * 16 + r + 8 * half;
        int hd = j * 16 + l16;
        int b = m >> 9, n = m & (N_ - 1);
        size_t bh = (size_t)(b * H_ + h);
        float val = c[i][j][r];
        if (which == 0)       Qs[(bh * N_ + n) * HD_ + hd] = f2bf(val * SCALE_);
        else if (which == 1)  Ks[(bh * N_ + n) * HD_ + hd] = f2bf(val);
        else                  Vt[(bh * HD_ + hd) * N_ + n] = f2bf(val);
      }
}

// ---------------------------------------------------------------------------
// Flash attention. All 8 waves of a block share one (b,h) and consume the
// same 32-key K/V tiles each iteration, so the tiles are staged cooperatively
// into LDS with double-buffered global_load_async_to_lds_b128 (ASYNCcnt),
// published with s_wait_asynccnt + workgroup barrier.
// Row bias mask[:, :, None] cancels in softmax; only the column bias applies.
__global__ __launch_bounds__(256) void attn_kernel(
    const bf16_t* __restrict__ Qs, const bf16_t* __restrict__ Ks,
    const bf16_t* __restrict__ Vt, const float* __restrict__ mask,
    bf16_t* __restrict__ AttnOut) {
  __shared__ bf16_t kvK[2][32 * HD_];             // key-tile: 32 rows x 64 hd
  __shared__ bf16_t kvV[2][HD_ * 32];             // v-tile:   64 hd  x 32 keys
  __shared__ bf16_t plds[8 * 16 * 32];            // 16x32 P tile per wave
  const int lane = threadIdx.x & 31, wave = threadIdx.x >> 5;
  const int half = lane >> 4, l16 = lane & 15;
  const int wg = blockIdx.x * 8 + wave;           // 0..16383
  const int qb = wg & 31;                         // query tile (16 rows)
  const int bh = wg >> 5;                         // b*16+h (block-uniform)
  const int b = bh >> 4, h = bh & (H_ - 1);

  // --- async staging addresses (per lane) ---
  // K tile: wave w copies rows 4w..4w+3 (128B each); lane -> (row, 16B chunk)
  const bf16_t* kg0 = Ks + ((size_t)bh * N_ + wave * 4 + (lane >> 3)) * HD_
                         + (lane & 7) * 8;
  // V tile: wave w copies hd-rows 8w..8w+7 (64B each of the 32-key window)
  const bf16_t* vg0 = Vt + ((size_t)bh * HD_ + wave * 8 + (lane >> 2)) * N_
                         + (lane & 3) * 8;
  unsigned klo[2], vlo[2];
#pragma unroll
  for (int p = 0; p < 2; ++p) {
    klo[p] = (unsigned)(size_t)&kvK[p][(wave * 4 + (lane >> 3)) * HD_ + (lane & 7) * 8];
    vlo[p] = (unsigned)(size_t)&kvV[p][(wave * 8 + (lane >> 2)) * 32 + (lane & 3) * 8];
  }

  const bf16_t* Qrow = Qs + ((size_t)bh * N_ + qb * 16 + l16) * HD_;
  v16bf qa0 = load_a_frag(Qrow + 0, half);
  v16bf qa1 = load_a_frag(Qrow + 32, half);

  v8f o[4];
#pragma unroll
  for (int t = 0; t < 4; ++t) o[t] = v8f{};
  float mrow[8], ssum[8];
#pragma unroll
  for (int r = 0; r < 8; ++r) { mrow[r] = -1e30f; ssum[r] = 0.f; }

  const float* maskb = mask + b * N_;
  bf16_t* pw = plds + wave * (16 * 32);

  // prologue: stage tile 0 into buffer 0
  async_copy_b128(klo[0], kg0);
  async_copy_b128(vlo[0], vg0);

  for (int kb = 0; kb < N_ / 32; ++kb) {
    const int cur = kb & 1;
    const int key0 = kb * 32;
    wait_async0();          // own copies landed
    __syncthreads();        // everyone's copies landed; prev readers done
    if (kb + 1 < N_ / 32) { // stage next tile into the other buffer
      async_copy_b128(klo[cur ^ 1], kg0 + (size_t)(kb + 1) * 32 * HD_);
      async_copy_b128(vlo[cur ^ 1], vg0 + (kb + 1) * 32);
    }

    const bf16_t* Kt = &kvK[cur][0];
    const bf16_t* Vtl = &kvV[cur][0];
    v8f s[2];
#pragma unroll
    for (int si = 0; si < 2; ++si) {
      const bf16_t* Krow = Kt + (si * 16 + l16) * HD_;
      v8f acc = v8f{};
      acc = wmma_bf16(qa0, load_b_frag(Krow + 0, half), acc);
      acc = wmma_bf16(qa1, load_b_frag(Krow + 32, half), acc);
      float mc = maskb[key0 + si * 16 + l16];     // column bias (per lane)
#pragma unroll
      for (int r = 0; r < 8; ++r) acc[r] += mc;
      s[si] = acc;
    }
    // online softmax update; lane-half-local 16-lane butterfly for row max
#pragma unroll
    for (int r = 0; r < 8; ++r) {
      float t = fmaxf(s[0][r], s[1][r]);
      t = fmaxf(t, __shfl_xor(t, 1, 32));
      t = fmaxf(t, __shfl_xor(t, 2, 32));
      t = fmaxf(t, __shfl_xor(t, 4, 32));
      t = fmaxf(t, __shfl_xor(t, 8, 32));
      float nm = fmaxf(mrow[r], t);
      float sc = EXP2F((mrow[r] - nm) * LOG2E_);
      mrow[r] = nm;
      float p0 = EXP2F((s[0][r] - nm) * LOG2E_);
      float p1 = EXP2F((s[1][r] - nm) * LOG2E_);
      ssum[r] = ssum[r] * sc + p0 + p1;           // per-lane partial row sum
      o[0][r] *= sc; o[1][r] *= sc; o[2][r] *= sc; o[3][r] *= sc;
      int m = r + 8 * half;                       // C/D layout -> LDS (m,k)
      pw[m * 32 + l16]      = f2bf(p0);
      pw[m * 32 + 16 + l16] = f2bf(p1);
    }
    wait_ds0();                                   // same-wave DS ordering
    v16bf pa = load_a_frag(pw + l16 * 32, half);  // P in A layout
#pragma unroll
    for (int t = 0; t < 4; ++t) {
      o[t] = wmma_bf16(pa, load_b_frag(Vtl + (t * 16 + l16) * 32, half), o[t]);
    }
  }

#pragma unroll
  for (int r = 0; r < 8; ++r) {
    float sum = ssum[r];
    sum += __shfl_xor(sum, 1, 32);
    sum += __shfl_xor(sum, 2, 32);
    sum += __shfl_xor(sum, 4, 32);
    sum += __shfl_xor(sum, 8, 32);
    float inv = 1.0f / sum;
    int q = qb * 16 + r + 8 * half;
    size_t base = ((size_t)b * N_ + q) * C_ + h * HD_ + l16;
    AttnOut[base +  0] = f2bf(o[0][r] * inv);
    AttnOut[base + 16] = f2bf(o[1][r] * inv);
    AttnOut[base + 32] = f2bf(o[2][r] * inv);
    AttnOut[base + 48] = f2bf(o[3][r] * inv);
  }
}

// ---------------------------------------------------------------------------
// Output projection: AttnOut[16384,1024] x W_proj^T + b_proj -> fp32 out.
__global__ __launch_bounds__(256) void proj_gemm_kernel(
    const bf16_t* __restrict__ Ao, const bf16_t* __restrict__ Wp,
    const float* __restrict__ bias, float* __restrict__ out) {
  const int lane = threadIdx.x & 31, wave = threadIdx.x >> 5;
  const int half = lane >> 4, l16 = lane & 15;
  const int wg = blockIdx.x * 8 + wave;           // 0..8191
  const int mt = wg / (C_ / 64), nt = wg % (C_ / 64);
  const int m0 = mt * 32, n0 = nt * 64;

  v8f c[2][4];
#pragma unroll
  for (int i = 0; i < 2; ++i)
#pragma unroll
    for (int j = 0; j < 4; ++j) c[i][j] = v8f{};

  const bf16_t* arow[2];
  arow[0] = Ao + (size_t)(m0 + l16) * C_;
  arow[1] = Ao + (size_t)(m0 + 16 + l16) * C_;
  const bf16_t* brow[4];
  float bj[4];
#pragma unroll
  for (int j = 0; j < 4; ++j) {
    brow[j] = Wp + (size_t)(n0 + j * 16 + l16) * C_;
    bj[j] = bias[n0 + j * 16 + l16];
  }

  for (int k0 = 0; k0 < C_; k0 += 32) {
    v16bf a0 = load_a_frag(arow[0] + k0, half);
    v16bf a1 = load_a_frag(arow[1] + k0, half);
#pragma unroll
    for (int j = 0; j < 4; ++j) {
      v16bf bb = load_b_frag(brow[j] + k0, half);
      c[0][j] = wmma_bf16(a0, bb, c[0][j]);
      c[1][j] = wmma_bf16(a1, bb, c[1][j]);
    }
  }

#pragma unroll
  for (int i = 0; i < 2; ++i)
#pragma unroll
    for (int j = 0; j < 4; ++j)
#pragma unroll
      for (int r = 0; r < 8; ++r) {
        int m = m0 + i * 16 + r + 8 * half;
        int d = n0 + j * 16 + l16;
        out[(size_t)m * C_ + d] = c[i][j][r] + bj[j];
      }
}

// ---------------------------------------------------------------------------
extern "C" void kernel_launch(void* const* d_in, const int* in_sizes, int n_in,
                              void* d_out, int out_size, void* d_ws, size_t ws_size,
                              hipStream_t stream) {
  (void)in_sizes; (void)n_in; (void)out_size; (void)ws_size;
  const float* x      = (const float*)d_in[0];   // [B,N,C]
  const float* mask   = (const float*)d_in[1];   // [B,N]
  const float* W_qkv  = (const float*)d_in[2];   // [3C,C]
  const float* W_proj = (const float*)d_in[3];   // [C,C]
  const float* b_proj = (const float*)d_in[4];   // [C]
  float* out = (float*)d_out;

  char* ws = (char*)d_ws;
  size_t off = 0;
  bf16_t* Xb  = (bf16_t*)(ws + off); off += (size_t)M_ * C_ * 2;        // 32 MB
  bf16_t* Wqb = (bf16_t*)(ws + off); off += (size_t)C3_ * C_ * 2;       //  6 MB
  bf16_t* Wpb = (bf16_t*)(ws + off); off += (size_t)C_ * C_ * 2;        //  2 MB
  bf16_t* Qs  = (bf16_t*)(ws + off); off += (size_t)M_ * C_ * 2;        // 32 MB
  bf16_t* Ks  = (bf16_t*)(ws + off); off += (size_t)M_ * C_ * 2;        // 32 MB
  bf16_t* Vt  = (bf16_t*)(ws + off); off += (size_t)M_ * C_ * 2;        // 32 MB
  bf16_t* Ao  = (bf16_t*)(ws + off); off += (size_t)M_ * C_ * 2;        // 32 MB

  const int nx = M_ * C_, nwq = C3_ * C_, nwp = C_ * C_;
  cvt_f32_bf16_kernel<<<(nx  + 255) / 256, 256, 0, stream>>>(x,      Xb,  nx);
  cvt_f32_bf16_kernel<<<(nwq + 255) / 256, 256, 0, stream>>>(W_qkv,  Wqb, nwq);
  cvt_f32_bf16_kernel<<<(nwp + 255) / 256, 256, 0, stream>>>(W_proj, Wpb, nwp);

  // (M/32)*(3C/64) = 512*48 = 24576 waves -> 3072 blocks of 8 waves
  qkv_gemm_kernel<<<3072, 256, 0, stream>>>(Xb, Wqb, Qs, Ks, Vt);
  // B*H*(N/16) = 16384 waves -> 2048 blocks
  attn_kernel<<<2048, 256, 0, stream>>>(Qs, Ks, Vt, mask, Ao);
  // (M/32)*(C/64) = 512*16 = 8192 waves -> 1024 blocks
  proj_gemm_kernel<<<1024, 256, 0, stream>>>(Ao, Wpb, b_proj, out);
}
